// Single_Head_58969900974493
// MI455X (gfx1250) — matrix operand order repaired and checked by reading, not compiled, not run on hardware
//
#include <hip/hip_runtime.h>

// ---------------------------------------------------------------------------
// Single-head causal attention, MI455X (gfx1250, wave32, WMMA f16->f32).
//   B=8, T=2048, E=1024, C=128.
// Kernel 1: fused QKV projection (f32 in -> f16 q/k/v in workspace, q
//           pre-scaled by C^-0.5), 128x128 tiles, v_wmma_f32_16x16x32_f16.
// Kernel 2: flash attention (online softmax, causal), 64 query rows per
//           block (4 waves x 16 rows), f32 accumulation, f32 output.
// ---------------------------------------------------------------------------

typedef _Float16 v8h  __attribute__((ext_vector_type(8)));
typedef _Float16 v16h __attribute__((ext_vector_type(16)));
typedef float    v8f  __attribute__((ext_vector_type(8)));

constexpr int kB  = 8;
constexpr int kT  = 2048;
constexpr int kE  = 1024;
constexpr int kC  = 128;
constexpr int kBT = kB * kT;   // 16384

static __device__ __forceinline__ v16h cat8(v8h lo, v8h hi) {
  return __builtin_shufflevector(lo, hi, 0, 1, 2, 3, 4, 5, 6, 7,
                                         8, 9, 10, 11, 12, 13, 14, 15);
}

static __device__ __forceinline__ v8f wmma_f16(v16h a, v16h b, v8f c) {
  // D = A(16x32 f16) * B(32x16 f16) + C(16x16 f32)
  return __builtin_amdgcn_wmma_f32_16x16x32_f16(false, a, false, b,
                                                (short)0, c, false, false);
}

// ===========================================================================
// Kernel 1: out[m, n] = sum_k x[m, k] * W[k, n]   (M=16384, N=128, K=1024)
// gridDim = (128, 3): blockIdx.y selects {Wk->kh, Wq->qh(scaled), Wv->vh}.
// 256 threads = 8 waves; wave (mg= w>>1 in 0..3, ng= w&1 in 0..1) computes a
// 32x64 sub-tile = 2x4 WMMA tiles.
// ===========================================================================
__global__ __launch_bounds__(256, 1) void qkv_proj_kernel(
    const float* __restrict__ x,
    const float* __restrict__ Wk, const float* __restrict__ Wq,
    const float* __restrict__ Wv,
    _Float16* __restrict__ kh, _Float16* __restrict__ qh,
    _Float16* __restrict__ vh) {
  const int mat = blockIdx.y;
  const float* __restrict__ W = (mat == 0) ? Wk : (mat == 1) ? Wq : Wv;
  _Float16* __restrict__ out  = (mat == 0) ? kh : (mat == 1) ? qh : vh;
  const float oscale = (mat == 1) ? 0.08838834764831845f : 1.0f;  // C^-0.5

  const int m0    = blockIdx.x * 128;
  const int tid   = threadIdx.x;
  const int lane  = tid & 31;
  const int w     = tid >> 5;
  const int mg    = w >> 1;        // 0..3  -> 32-row group
  const int ng    = w & 1;         // 0..1  -> 64-col group
  const int lmod  = lane & 15;
  const int lhalf = lane >> 4;

  // LDS staging, padded stride 40 f16 (80 B: 16B-aligned, bank-spread)
  __shared__ _Float16 xs[128][40];  // x tile, row-major [m][k]
  __shared__ _Float16 ws[128][40];  // W tile, TRANSPOSED [n][k] (B-frag layout)

  v8f acc[2][4];
#pragma unroll
  for (int mi = 0; mi < 2; ++mi)
#pragma unroll
    for (int nj = 0; nj < 4; ++nj) acc[mi][nj] = (v8f)0.0f;

  for (int kk = 0; kk < kE; kk += 32) {
    __syncthreads();
    // stage x tile: 128 rows x 32 k (f32 -> f16), 8 float4 per row
    for (int i = tid; i < 128 * 8; i += 256) {
      const int r = i >> 3, c4 = (i & 7) * 4;
      const float4 f =
          *(const float4*)(x + (size_t)(m0 + r) * kE + kk + c4);
      _Float16* p = &xs[r][c4];
      p[0] = (_Float16)f.x; p[1] = (_Float16)f.y;
      p[2] = (_Float16)f.z; p[3] = (_Float16)f.w;
    }
    // stage W tile transposed: 32 k-rows x 128 n (f32 -> f16)
    for (int i = tid; i < 32 * 32; i += 256) {
      const int r = i >> 5, c4 = (i & 31) * 4;
      const float4 f = *(const float4*)(W + (size_t)(kk + r) * kC + c4);
      ws[c4 + 0][r] = (_Float16)f.x; ws[c4 + 1][r] = (_Float16)f.y;
      ws[c4 + 2][r] = (_Float16)f.z; ws[c4 + 3][r] = (_Float16)f.w;
    }
    __syncthreads();

    // A fragments: lane = row lmod; K = lhalf*8+{0..7}, 16+lhalf*8+{0..7}
    v16h a[2];
#pragma unroll
    for (int mi = 0; mi < 2; ++mi) {
      const int row = mg * 32 + mi * 16 + lmod;
      const v8h lo = *(const v8h*)&xs[row][lhalf * 8];
      const v8h hi = *(const v8h*)&xs[row][16 + lhalf * 8];
      a[mi] = cat8(lo, hi);
    }
    // B fragments: lane = col lmod; K = lhalf*8+{0..7}, 16+lhalf*8+{0..7}
#pragma unroll
    for (int nj = 0; nj < 4; ++nj) {
      const int n = ng * 64 + nj * 16 + lmod;
      const v8h lo = *(const v8h*)&ws[n][lhalf * 8];
      const v8h hi = *(const v8h*)&ws[n][16 + lhalf * 8];
      const v16h b = cat8(lo, hi);
      acc[0][nj] = wmma_f16(a[0], b, acc[0][nj]);
      acc[1][nj] = wmma_f16(a[1], b, acc[1][nj]);
    }
  }

  // store f16 (D layout: VGPR r -> row r + 8*lhalf, col = lmod)
#pragma unroll
  for (int mi = 0; mi < 2; ++mi)
#pragma unroll
    for (int nj = 0; nj < 4; ++nj)
#pragma unroll
      for (int r = 0; r < 8; ++r) {
        const int row = m0 + mg * 32 + mi * 16 + lhalf * 8 + r;
        const int col = ng * 64 + nj * 16 + lmod;
        out[(size_t)row * kC + col] = (_Float16)(acc[mi][nj][r] * oscale);
      }
}

// ===========================================================================
// Kernel 2: flash attention. grid = (T/64, B), 128 threads = 4 waves.
// Wave w owns rows [q0 + 16w, q0 + 16w + 16); Bc = 64 key rows per step.
// ===========================================================================
__global__ __launch_bounds__(128, 1) void flash_attn_kernel(
    const _Float16* __restrict__ qh, const _Float16* __restrict__ kh,
    const _Float16* __restrict__ vh, float* __restrict__ out) {
  const int b     = blockIdx.y;
  const int q0    = blockIdx.x * 64;
  const int tid   = threadIdx.x;
  const int lane  = tid & 31;
  const int w     = tid >> 5;      // 0..3
  const int lmod  = lane & 15;
  const int lhalf = lane >> 4;
  const int row0  = q0 + w * 16;

  __shared__ _Float16 qs[64][136];   // q tile row-major [m][c]
  __shared__ _Float16 ks[64][136];   // k tile row-major [s][c] (== B^T for QK^T)
  __shared__ _Float16 vs[128][72];   // v tile TRANSPOSED [c][s] (B^T for P@V)
  __shared__ _Float16 ps[4][16][72]; // per-wave P (D-layout -> A-layout relay)

  const _Float16* __restrict__ qb = qh + (size_t)b * kT * kC;
  const _Float16* __restrict__ kb = kh + (size_t)b * kT * kC;
  const _Float16* __restrict__ vb = vh + (size_t)b * kT * kC;

  // stage q tile once (16 x v8h per row)
  for (int i = tid; i < 64 * 16; i += 128) {
    const int r = i >> 4, c = (i & 15) * 8;
    *(v8h*)&qs[r][c] = *(const v8h*)(qb + (size_t)(q0 + r) * kC + c);
  }
  __syncthreads();

  // Hoist q A-fragments for all 4 K-steps (C = 128 = 4 x 32)
  v16h qa[4];
#pragma unroll
  for (int kk = 0; kk < 4; ++kk) {
    const int row = w * 16 + lmod;
    const v8h lo = *(const v8h*)&qs[row][kk * 32 + lhalf * 8];
    const v8h hi = *(const v8h*)&qs[row][kk * 32 + 16 + lhalf * 8];
    qa[kk] = cat8(lo, hi);
  }

  v8f  o[8];
  float mrow[8], lrow[8];
#pragma unroll
  for (int n = 0; n < 8; ++n) o[n] = (v8f)0.0f;
#pragma unroll
  for (int r = 0; r < 8; ++r) { mrow[r] = -1e30f; lrow[r] = 0.0f; }

  for (int s0 = 0; s0 <= q0; s0 += 64) {
    __syncthreads();
    // stage K tile (row-major)
    for (int i = tid; i < 64 * 16; i += 128) {
      const int r = i >> 4, c = (i & 15) * 8;
      *(v8h*)&ks[r][c] = *(const v8h*)(kb + (size_t)(s0 + r) * kC + c);
    }
    // stage V tile transposed: vs[c][s] = v[s0+s][c]
    for (int i = tid; i < 64 * 64; i += 128) {
      const int s = i >> 6, c = (i & 63) * 2;
      const unsigned u = *(const unsigned*)(vb + (size_t)(s0 + s) * kC + c);
      vs[c][s]     = ((const _Float16*)&u)[0];
      vs[c + 1][s] = ((const _Float16*)&u)[1];
    }
    __syncthreads();

    if (s0 <= row0 + 15) {  // tile not fully above this wave's diagonal
      // ---- S = q . k^T  (16 x 64, f32 accum) ----
      v8f S[4];
#pragma unroll
      for (int nj = 0; nj < 4; ++nj) S[nj] = (v8f)0.0f;
#pragma unroll
      for (int kk = 0; kk < 4; ++kk) {
#pragma unroll
        for (int nj = 0; nj < 4; ++nj) {
          const int n = nj * 16 + lmod;
          const v8h lo = *(const v8h*)&ks[n][kk * 32 + lhalf * 8];
          const v8h hi = *(const v8h*)&ks[n][kk * 32 + 16 + lhalf * 8];
          S[nj] = wmma_f16(qa[kk], cat8(lo, hi), S[nj]);
        }
      }
      // ---- causal mask ----
      if (s0 + 63 > row0) {
#pragma unroll
        for (int nj = 0; nj < 4; ++nj) {
          const int col = s0 + nj * 16 + lmod;
#pragma unroll
          for (int r = 0; r < 8; ++r) {
            const int row = row0 + lhalf * 8 + r;
            if (col > row) S[nj][r] = -1e30f;
          }
        }
      }
      // ---- online softmax (per row r; rows live on 16-lane halves) ----
      float scl[8];
#pragma unroll
      for (int r = 0; r < 8; ++r) {
        float mx = fmaxf(fmaxf(S[0][r], S[1][r]), fmaxf(S[2][r], S[3][r]));
        mx = fmaxf(mx, __shfl_xor(mx, 1, 32));
        mx = fmaxf(mx, __shfl_xor(mx, 2, 32));
        mx = fmaxf(mx, __shfl_xor(mx, 4, 32));
        mx = fmaxf(mx, __shfl_xor(mx, 8, 32));
        const float mn = fmaxf(mrow[r], mx);
        const float sc = __expf(mrow[r] - mn);
        mrow[r] = mn;
        float rs = 0.0f;
#pragma unroll
        for (int nj = 0; nj < 4; ++nj) {
          const float p = __expf(S[nj][r] - mn);
          S[nj][r] = p;
          rs += p;
        }
        rs += __shfl_xor(rs, 1, 32);
        rs += __shfl_xor(rs, 2, 32);
        rs += __shfl_xor(rs, 4, 32);
        rs += __shfl_xor(rs, 8, 32);
        lrow[r] = lrow[r] * sc + rs;
        scl[r] = sc;
      }
#pragma unroll
      for (int n = 0; n < 8; ++n)
#pragma unroll
        for (int r = 0; r < 8; ++r) o[n][r] *= scl[r];

      // ---- P: D-layout -> LDS -> A-layout (per-wave region, same-wave DS
      //      ops are in-order) ----
#pragma unroll
      for (int nj = 0; nj < 4; ++nj)
#pragma unroll
        for (int r = 0; r < 8; ++r)
          ps[w][lhalf * 8 + r][nj * 16 + lmod] = (_Float16)S[nj][r];

      // ---- o += P @ V  (K = 64 = 2 x 32) ----
#pragma unroll
      for (int kk = 0; kk < 2; ++kk) {
        const v8h plo = *(const v8h*)&ps[w][lmod][kk * 32 + lhalf * 8];
        const v8h phi = *(const v8h*)&ps[w][lmod][kk * 32 + 16 + lhalf * 8];
        const v16h pa = cat8(plo, phi);
#pragma unroll
        for (int n = 0; n < 8; ++n) {
          const int c = n * 16 + lmod;
          const v8h blo = *(const v8h*)&vs[c][kk * 32 + lhalf * 8];
          const v8h bhi = *(const v8h*)&vs[c][kk * 32 + 16 + lhalf * 8];
          o[n] = wmma_f16(pa, cat8(blo, bhi), o[n]);
        }
      }
    }
  }

  // ---- epilogue: normalize and store f32 ----
  float inv[8];
#pragma unroll
  for (int r = 0; r < 8; ++r) inv[r] = 1.0f / lrow[r];
#pragma unroll
  for (int n = 0; n < 8; ++n)
#pragma unroll
    for (int r = 0; r < 8; ++r) {
      const int row = row0 + lhalf * 8 + r;
      const int col = n * 16 + lmod;
      out[((size_t)b * kT + row) * kC + col] = o[n][r] * inv[r];
    }
}

// ===========================================================================
extern "C" void kernel_launch(void* const* d_in, const int* in_sizes, int n_in,
                              void* d_out, int out_size, void* d_ws,
                              size_t ws_size, hipStream_t stream) {
  const float* x  = (const float*)d_in[0];
  const float* Wk = (const float*)d_in[1];
  const float* Wq = (const float*)d_in[2];
  const float* Wv = (const float*)d_in[3];
  float* out = (float*)d_out;

  // workspace: f16 k, q, v  [B*T, C] each (4 MB each, 12 MB total)
  _Float16* kh = (_Float16*)d_ws;
  _Float16* qh = kh + (size_t)kBT * kC;
  _Float16* vh = qh + (size_t)kBT * kC;

  dim3 g1(kBT / 128, 3);
  qkv_proj_kernel<<<g1, 256, 0, stream>>>(x, Wk, Wq, Wv, kh, qh, vh);

  dim3 g2(kT / 64, kB);
  flash_attn_kernel<<<g2, 128, 0, stream>>>(qh, kh, vh, out);
}
